// SpatialAttention_18648747999777
// MI455X (gfx1250) — matrix operand order repaired and checked by reading, not compiled, not run on hardware
//
#include <hip/hip_runtime.h>

typedef __attribute__((ext_vector_type(16))) _Float16 v16h;
typedef __attribute__((ext_vector_type(8)))  _Float16 v8h;
typedef __attribute__((ext_vector_type(8)))  float    v8f;
typedef __attribute__((ext_vector_type(4)))  float    v4f;

#define NDIM 256
#define CDIM 256
#define LSTR 264   // padded LDS row stride in f16 (528 B, 16B aligned)
#define MID  128
#define EPSV 1e-12f

union AFrag { v16h v; v8h h[2]; };

__launch_bounds__(256)
__global__ void spatial_attn_kernel(const float* __restrict__ x,
                                    const float* __restrict__ kern,
                                    const float* __restrict__ bias,
                                    float* __restrict__ out)
{
    __shared__ _Float16 Xf[NDIM * LSTR];   // 135168 B
    __shared__ float simmid[NDIM];
    __shared__ float coe0[NDIM];
    __shared__ float sumsq[NDIM];
    __shared__ float dotv[NDIM];
    __shared__ float sarr[NDIM];
    __shared__ float red[8];
    __shared__ float scal0, scal1;

    const int tid  = threadIdx.x;
    const int lane = tid & 31;
    const int wave = tid >> 5;
    const int lm   = lane & 15;   // lane within 16-group
    const int kh   = lane >> 4;   // which half of the wave
    const int b    = blockIdx.x;

    const float* xb = x   + (size_t)b * NDIM * CDIM;
    float*       ob = out + (size_t)b * NDIM * CDIM;

    // ---- Phase 1: stage x[b] -> LDS as f16 (coalesced float4 reads) ----
    {
        const v4f* x4 = (const v4f*)xb;
        for (int idx = tid; idx < NDIM * CDIM / 4; idx += 256) {
            v4f v = x4[idx];
            int row  = idx >> 6;      // 64 float4 per row
            int col4 = idx & 63;
            _Float16* dst = &Xf[row * LSTR + col4 * 4];
            dst[0] = (_Float16)v.x; dst[1] = (_Float16)v.y;
            dst[2] = (_Float16)v.z; dst[3] = (_Float16)v.w;
        }
    }
    __syncthreads();

    // ---- Phase 2: sim[mid,:] (GEMV, b128-vectorized), norm_mid, coe0 ----
    {
        float acc = 0.f;
        const v8h* rm = (const v8h*)&Xf[MID * LSTR];
        const v8h* rt = (const v8h*)&Xf[tid * LSTR];
        for (int c8 = 0; c8 < CDIM / 8; ++c8) {
            v8h m8 = rm[c8];
            v8h t8 = rt[c8];
            #pragma unroll
            for (int e = 0; e < 8; ++e)
                acc += (float)m8[e] * (float)t8[e];
        }
        simmid[tid] = acc;
        float v = acc * acc;
        for (int s = 16; s >= 1; s >>= 1) v += __shfl_xor(v, s, 32);
        if (lane == 0) red[wave] = v;
    }
    __syncthreads();
    if (tid == 0) {
        float t = 0.f;
        for (int i = 0; i < 8; ++i) t += red[i];
        scal0 = rsqrtf(fmaxf(t, EPSV));
    }
    __syncthreads();
    coe0[tid] = scal0 * simmid[tid] * kern[tid];
    __syncthreads();

    // ---- Phase 3: WMMA sim tiles; fold sum(sim^2) and sum(sim*coe0) ----
    // Wave w owns tile rows {2w, 2w+1}; both rows live at once so every
    // B-fragment load feeds TWO wmmas (1 ds_load_b128 per wmma).
    // B is explicitly double-buffered so the dscnt wait is partial.
    {
        const int i0 = wave * 2;
        const int i1 = i0 + 1;
        AFrag a0[8], a1[8];
        const _Float16* ar0 = &Xf[(i0 * 16 + lm) * LSTR];
        const _Float16* ar1 = &Xf[(i1 * 16 + lm) * LSTR];
        #pragma unroll
        for (int kb = 0; kb < 8; ++kb) {
            a0[kb].h[0] = *(const v8h*)&ar0[kb * 32 + kh * 16];
            a0[kb].h[1] = *(const v8h*)&ar0[kb * 32 + kh * 16 + 8];
            a1[kb].h[0] = *(const v8h*)&ar1[kb * 32 + kh * 16];
            a1[kb].h[1] = *(const v8h*)&ar1[kb * 32 + kh * 16 + 8];
        }
        float sq0[8], dv0[8], sq1[8], dv1[8];
        #pragma unroll
        for (int r = 0; r < 8; ++r) { sq0[r] = dv0[r] = sq1[r] = dv1[r] = 0.f; }

        const _Float16* brow = &Xf[lm * LSTR];       // j = 0 row
        AFrag bcur, bnext;
        bcur.h[0] = *(const v8h*)&brow[kh * 16];
        bcur.h[1] = *(const v8h*)&brow[kh * 16 + 8];

        for (int j = 0; j < 16; ++j) {
            const float cv = coe0[j * 16 + lm];      // in flight behind wmmas
            const _Float16* nrow = &Xf[((j + 1) & 15) * 16 * LSTR + lm * LSTR];
            v8f c0 = {}, c1 = {};
            #pragma unroll
            for (int kb = 0; kb < 8; ++kb) {
                if (kb < 7) {
                    bnext.h[0] = *(const v8h*)&brow[(kb + 1) * 32 + kh * 16];
                    bnext.h[1] = *(const v8h*)&brow[(kb + 1) * 32 + kh * 16 + 8];
                } else {
                    // prefetch next j's first fragment
                    bnext.h[0] = *(const v8h*)&nrow[kh * 16];
                    bnext.h[1] = *(const v8h*)&nrow[kh * 16 + 8];
                }
                c0 = __builtin_amdgcn_wmma_f32_16x16x32_f16(
                        false, a0[kb].v, false, bcur.v, (short)0, c0, false, false);
                c1 = __builtin_amdgcn_wmma_f32_16x16x32_f16(
                        false, a1[kb].v, false, bcur.v, (short)0, c1, false, false);
                bcur = bnext;
            }
            brow = nrow;
            #pragma unroll
            for (int r = 0; r < 8; ++r) {
                sq0[r] += c0[r] * c0[r];
                dv0[r] += c0[r] * cv;
                sq1[r] += c1[r] * c1[r];
                dv1[r] += c1[r] * cv;
            }
        }
        // Reduce over the 16 lanes of each half-wave (columns of the tile).
        #pragma unroll
        for (int s = 1; s <= 8; s <<= 1) {
            #pragma unroll
            for (int r = 0; r < 8; ++r) {
                sq0[r] += __shfl_xor(sq0[r], s, 16);
                dv0[r] += __shfl_xor(dv0[r], s, 16);
                sq1[r] += __shfl_xor(sq1[r], s, 16);
                dv1[r] += __shfl_xor(dv1[r], s, 16);
            }
        }
        if (lm == 0) {
            const int b0 = i0 * 16 + kh * 8;   // lanes 0..15: M=r ; 16..31: M=8+r
            const int b1 = i1 * 16 + kh * 8;
            #pragma unroll
            for (int r = 0; r < 8; ++r) {
                sumsq[b0 + r] = sq0[r];
                dotv[b0 + r]  = dv0[r];
                sumsq[b1 + r] = sq1[r];
                dotv[b1 + r]  = dv1[r];
            }
        }
    }
    __syncthreads();

    // ---- Phase 4: w, softmax over N ----
    float w = rsqrtf(fmaxf(sumsq[tid], EPSV)) * dotv[tid] + bias[tid];
    {
        float m = w;
        for (int s = 16; s >= 1; s >>= 1) m = fmaxf(m, __shfl_xor(m, s, 32));
        if (lane == 0) red[wave] = m;
    }
    __syncthreads();
    if (tid == 0) {
        float t = red[0];
        for (int i = 1; i < 8; ++i) t = fmaxf(t, red[i]);
        scal0 = t;
    }
    __syncthreads();
    const float p = __expf(w - scal0);
    {
        float sm = p;
        for (int s = 16; s >= 1; s >>= 1) sm += __shfl_xor(sm, s, 32);
        if (lane == 0) red[wave] = sm;
    }
    __syncthreads();
    if (tid == 0) {
        float t = 0.f;
        for (int i = 0; i < 8; ++i) t += red[i];
        scal1 = t;
    }
    __syncthreads();
    sarr[tid] = p / scal1;
    __syncthreads();

    // ---- Phase 5: out = softmax * x (coalesced float4, NT store) ----
    {
        const v4f* x4 = (const v4f*)xb;
        v4f*       o4 = (v4f*)ob;
        for (int idx = tid; idx < NDIM * CDIM / 4; idx += 256) {
            v4f v = __builtin_nontemporal_load(&x4[idx]);
            const float s = sarr[idx >> 6];
            v *= s;
            __builtin_nontemporal_store(v, &o4[idx]);
        }
    }
}

extern "C" void kernel_launch(void* const* d_in, const int* in_sizes, int n_in,
                              void* d_out, int out_size, void* d_ws, size_t ws_size,
                              hipStream_t stream) {
    const float* x    = (const float*)d_in[0];   // [512,256,256]
    const float* kern = (const float*)d_in[1];   // [256,1]
    const float* bias = (const float*)d_in[2];   // [256]
    float* out = (float*)d_out;                  // [512,256,256]
    (void)in_sizes; (void)n_in; (void)out_size; (void)d_ws; (void)ws_size;
    spatial_attn_kernel<<<512, 256, 0, stream>>>(x, kern, bias, out);
}
